// WindowAttention_18494129176747
// MI455X (gfx1250) — matrix operand order repaired and checked by reading, not compiled, not run on hardware
//
#include <hip/hip_runtime.h>

typedef __bf16 bf16_t;
typedef __attribute__((ext_vector_type(16))) __bf16 v16bf;
typedef __attribute__((ext_vector_type(8)))  __bf16 v8bf;
typedef __attribute__((ext_vector_type(8)))  float  v8f;
typedef __attribute__((ext_vector_type(4)))  float  f4v;

#define DIM 256
#define NTOK 64
#define HEADS 8
#define HD 32
#define QKV_OC 768            // 256 q + 256 k + 256 v
#define SMEM_BYTES (136 * 1024)

__device__ __forceinline__ v16bf cat8(v8bf lo, v8bf hi) {
    return __builtin_shufflevector(lo, hi, 0,1,2,3,4,5,6,7,8,9,10,11,12,13,14,15);
}

__device__ __forceinline__ v8f wmma_bf16(v16bf a, v16bf b, v8f c) {
    // D = A(16x32 bf16) * B(32x16 bf16) + C(16x16 f32)
    return __builtin_amdgcn_wmma_f32_16x16x32_bf16(
        false, a, false, b, (short)0, c, false, false);
}

// ---------------------------------------------------------------------------
// Pre-pass: convert Wq|Wkv -> WtAll[oc][ic] bf16 (column-major-as-B), and
// Wp -> WpT[oc][ic] bf16. Weights stay L2-resident (only ~0.5 MB total).
// ---------------------------------------------------------------------------
__global__ void winattn_convert_weights(const float* __restrict__ Wq,
                                        const float* __restrict__ Wkv,
                                        const float* __restrict__ Wp,
                                        bf16_t* __restrict__ WtAll,
                                        bf16_t* __restrict__ WpT) {
    int i = blockIdx.x * blockDim.x + threadIdx.x;
    const int N1 = QKV_OC * DIM;          // 196608
    if (i < N1) {
        int oc = i >> 8, ic = i & 255;
        float v = (oc < DIM) ? Wq[ic * DIM + oc] : Wkv[ic * 2 * DIM + (oc - DIM)];
        WtAll[i] = (bf16_t)v;
    } else if (i < N1 + DIM * DIM) {
        int j = i - N1;
        int oc = j >> 8, ic = j & 255;
        WpT[j] = (bf16_t)Wp[ic * DIM + oc];
    }
}

// ---------------------------------------------------------------------------
// Fused window attention: one workgroup (8 waves) per window; wave h owns head h.
// LDS map (dynamic, 136 KB):
//   [  0K, 32K) region0 : xs bf16[64][256]  -> later O bf16[64][256]
//   [ 32K, 96K) qk      : q bf16[64][256] | k bf16[64][256]
//                          -> later P bf16[8][64][64] -> later outf f32[64][256]
//   [ 96K,128K) vT      : bf16[256][64]   (v transposed: B-operand for P@V)
//   [128K,136K) biasT   : f32[225*8]
// ---------------------------------------------------------------------------
__global__ void __launch_bounds__(256, 2)
winattn_fused(const float* __restrict__ x,
              const float* __restrict__ bq,
              const float* __restrict__ bkv,
              const float* __restrict__ bias_table,
              const bf16_t* __restrict__ WtAll,
              const bf16_t* __restrict__ WpT,
              const float* __restrict__ bp,
              float* __restrict__ out) {
    extern __shared__ __align__(16) char smem_raw[];
    bf16_t* xs    = (bf16_t*)(smem_raw);                 // 32 KB
    bf16_t* qbuf  = (bf16_t*)(smem_raw + 32 * 1024);     // 32 KB
    bf16_t* kbuf  = (bf16_t*)(smem_raw + 64 * 1024);     // 32 KB
    bf16_t* vT    = (bf16_t*)(smem_raw + 96 * 1024);     // 32 KB
    float*  biasT = (float*)(smem_raw + 128 * 1024);     //  8 KB
    bf16_t* pbuf  = qbuf;                                // alias (64 KB)
    bf16_t* obuf  = xs;                                  // alias (32 KB)
    float*  outf  = (float*)qbuf;                        // alias (64 KB)

    const int tid  = threadIdx.x;
    const int wave = tid >> 5;
    const int lane = tid & 31;
    const int ln   = lane & 15;
    const int hi   = lane >> 4;                          // lane half: 0 / 1
    const int b    = blockIdx.x;

    // ---- stage x tile (f32 -> bf16 LDS) + bias table, coalesced b128 -------
    {
        const f4v* xb = (const f4v*)(x + (size_t)b * (NTOK * DIM));
        #pragma unroll
        for (int j = 0; j < 16; ++j) {
            int i4 = tid + j * 256;                      // 4096 float4 total
            f4v v = xb[i4];
            int base = i4 * 4;
            xs[base + 0] = (bf16_t)v[0];
            xs[base + 1] = (bf16_t)v[1];
            xs[base + 2] = (bf16_t)v[2];
            xs[base + 3] = (bf16_t)v[3];
        }
        for (int j = tid; j < 225 * HEADS; j += 256) biasT[j] = bias_table[j];
    }
    __syncthreads();

    // ---- Phase 1: [64,256] @ [256,768] -> q (scaled), k, vT ----------------
    {
        #pragma unroll
        for (int i = 0; i < 6; ++i) {
            const int nT = wave * 6 + i;                 // 0..47
            const int oc = nT * 16 + ln;
            const float bia = (oc < DIM) ? bq[oc] : bkv[oc - DIM];
            v8f acc[4] = {};
            #pragma unroll
            for (int k = 0; k < 8; ++k) {
                v16bf bfrag = *(const v16bf*)(WtAll + oc * DIM + k * 32 + hi * 16);
                const int kb = k * 32 + hi * 16;
                #pragma unroll
                for (int m = 0; m < 4; ++m) {
                    const int row = m * 16 + ln;
                    v8bf lo = *(const v8bf*)(xs + row * DIM + kb);
                    v8bf hh = *(const v8bf*)(xs + row * DIM + kb + 8);
                    acc[m] = wmma_bf16(cat8(lo, hh), bfrag, acc[m]);
                }
            }
            #pragma unroll
            for (int m = 0; m < 4; ++m) {
                #pragma unroll
                for (int r = 0; r < 8; ++r) {
                    const int n = m * 16 + hi * 8 + r;   // D layout: lane=N, vgpr=M%8
                    float v = acc[m][r] + bia;
                    if (oc < 256)      qbuf[n * DIM + oc]              = (bf16_t)(v * 0.17677669529663689f);
                    else if (oc < 512) kbuf[n * DIM + (oc - 256)]      = (bf16_t)v;
                    else               vT[(oc - 512) * NTOK + n]       = (bf16_t)v;
                }
            }
        }
    }
    __syncthreads();

    // ---- Phase 2: per-head attention; wave h == head h ---------------------
    const int h = wave;
    v8f s[4][4] = {};
    {
        v16bf bk[4];
        #pragma unroll
        for (int nT = 0; nT < 4; ++nT) {
            const int km = nT * 16 + ln;                 // B: lane = N(=key m)
            bk[nT] = *(const v16bf*)(kbuf + km * DIM + h * HD + hi * 16);
        }
        #pragma unroll
        for (int mT = 0; mT < 4; ++mT) {
            const int row = mT * 16 + ln;
            const int kb  = h * HD + hi * 16;
            v8bf lo = *(const v8bf*)(qbuf + row * DIM + kb);
            v8bf hh = *(const v8bf*)(qbuf + row * DIM + kb + 8);
            v16bf a = cat8(lo, hh);
            #pragma unroll
            for (int nT = 0; nT < 4; ++nT)
                s[mT][nT] = wmma_bf16(a, bk[nT], s[mT][nT]);
        }
    }
    // relative-position bias + row softmax (rows live in one 16-lane half)
    #pragma unroll
    for (int mT = 0; mT < 4; ++mT) {
        #pragma unroll
        for (int r = 0; r < 8; ++r) {
            const int n = mT * 16 + hi * 8 + r;
            const int chn = n >> 3, cwn = n & 7;
            float vals[4];
            float rmax = -3.0e38f;
            #pragma unroll
            for (int nT = 0; nT < 4; ++nT) {
                const int m = nT * 16 + ln;
                const int idx = (chn - (m >> 3) + 7) * 15 + (cwn - (m & 7) + 7);
                float v = s[mT][nT][r] + biasT[idx * HEADS + h];
                vals[nT] = v;
                rmax = fmaxf(rmax, v);
            }
            #pragma unroll
            for (int off = 1; off < 16; off <<= 1)
                rmax = fmaxf(rmax, __shfl_xor(rmax, off, 32));
            float rsum = 0.f;
            #pragma unroll
            for (int nT = 0; nT < 4; ++nT) { vals[nT] = __expf(vals[nT] - rmax); rsum += vals[nT]; }
            #pragma unroll
            for (int off = 1; off < 16; off <<= 1)
                rsum += __shfl_xor(rsum, off, 32);
            const float inv = 1.0f / rsum;
            #pragma unroll
            for (int nT = 0; nT < 4; ++nT) s[mT][nT][r] = vals[nT] * inv;
        }
    }
    __syncthreads();   // q/k reads done by ALL waves -> safe to overwrite with P

    bf16_t* pw = pbuf + h * (NTOK * NTOK);
    #pragma unroll
    for (int mT = 0; mT < 4; ++mT)
        #pragma unroll
        for (int r = 0; r < 8; ++r) {
            const int n = mT * 16 + hi * 8 + r;
            #pragma unroll
            for (int nT = 0; nT < 4; ++nT)
                pw[n * NTOK + nT * 16 + ln] = (bf16_t)s[mT][nT][r];
        }

    // ---- Phase 3: O_h = P @ V_h  ([64,64] @ [64,32]) -----------------------
    {
        v8f o_[4][2] = {};
        #pragma unroll
        for (int mT = 0; mT < 4; ++mT) {
            #pragma unroll
            for (int ks = 0; ks < 2; ++ks) {
                const int row = mT * 16 + ln;
                const int kb  = ks * 32 + hi * 16;
                v8bf lo = *(const v8bf*)(pw + row * NTOK + kb);
                v8bf hh = *(const v8bf*)(pw + row * NTOK + kb + 8);
                v16bf a = cat8(lo, hh);
                #pragma unroll
                for (int nT = 0; nT < 2; ++nT) {
                    const int d = h * HD + nT * 16 + ln;
                    v16bf bb = *(const v16bf*)(vT + d * NTOK + ks * 32 + hi * 16);
                    o_[mT][nT] = wmma_bf16(a, bb, o_[mT][nT]);
                }
            }
        }
        #pragma unroll
        for (int mT = 0; mT < 4; ++mT)
            #pragma unroll
            for (int nT = 0; nT < 2; ++nT)
                #pragma unroll
                for (int r = 0; r < 8; ++r) {
                    const int n = mT * 16 + hi * 8 + r;
                    obuf[n * DIM + h * HD + nT * 16 + ln] = (bf16_t)o_[mT][nT][r];
                }
    }
    __syncthreads();

    // ---- Phase 4: out = O @ Wp^T + bp -> f32 staging in LDS ----------------
    #pragma unroll
    for (int t = 0; t < 2; ++t) {
        const int nT4 = wave * 2 + t;
        const int oc  = nT4 * 16 + ln;
        const float bia = bp[oc];
        v8f f[4] = {};
        #pragma unroll
        for (int k = 0; k < 8; ++k) {
            v16bf bfrag = *(const v16bf*)(WpT + oc * DIM + k * 32 + hi * 16);
            const int kb = k * 32 + hi * 16;
            #pragma unroll
            for (int mT = 0; mT < 4; ++mT) {
                const int row = mT * 16 + ln;
                v8bf lo = *(const v8bf*)(obuf + row * DIM + kb);
                v8bf hh = *(const v8bf*)(obuf + row * DIM + kb + 8);
                f[mT] = wmma_bf16(cat8(lo, hh), bfrag, f[mT]);
            }
        }
        #pragma unroll
        for (int mT = 0; mT < 4; ++mT)
            #pragma unroll
            for (int r = 0; r < 8; ++r)
                outf[(mT * 16 + hi * 8 + r) * DIM + oc] = f[mT][r] + bia;
    }
    __syncthreads();

    // ---- coalesced b128 writeback ------------------------------------------
    {
        const f4v* src = (const f4v*)outf;
        f4v* dst = (f4v*)(out + (size_t)b * (NTOK * DIM));
        #pragma unroll
        for (int j = 0; j < 16; ++j)
            dst[tid + j * 256] = src[tid + j * 256];
    }
}

// ---------------------------------------------------------------------------
extern "C" void kernel_launch(void* const* d_in, const int* in_sizes, int n_in,
                              void* d_out, int out_size, void* d_ws, size_t ws_size,
                              hipStream_t stream) {
    (void)in_sizes; (void)n_in; (void)out_size; (void)ws_size;
    const float* x          = (const float*)d_in[0];
    const float* Wq         = (const float*)d_in[1];
    const float* bq         = (const float*)d_in[2];
    const float* Wkv        = (const float*)d_in[3];
    const float* bkv        = (const float*)d_in[4];
    const float* bias_table = (const float*)d_in[5];
    const float* Wp         = (const float*)d_in[6];
    const float* bp         = (const float*)d_in[7];
    float* out = (float*)d_out;

    bf16_t* WtAll = (bf16_t*)d_ws;                                   // 768*256*2 B
    bf16_t* WpT   = (bf16_t*)((char*)d_ws + QKV_OC * DIM * 2);       // 256*256*2 B

    const int convN = QKV_OC * DIM + DIM * DIM;                      // 262144
    winattn_convert_weights<<<(convN + 255) / 256, 256, 0, stream>>>(
        Wq, Wkv, Wp, WtAll, WpT);

    (void)hipFuncSetAttribute((const void*)winattn_fused,
                              hipFuncAttributeMaxDynamicSharedMemorySize,
                              SMEM_BYTES);

    const int B = 4096;
    winattn_fused<<<B, 256, SMEM_BYTES, stream>>>(
        x, bq, bkv, bias_table, WtAll, WpT, bp, out);
}
